// MultiHeadAttention_7937099563103
// MI455X (gfx1250) — matrix operand order repaired and checked by reading, not compiled, not run on hardware
//
#include <hip/hip_runtime.h>
#include <hip/hip_bf16.h>
#include <math.h>

// ---------------------------------------------------------------------------
// Types for CDNA5 WMMA (wave32, v_wmma_f32_16x16x32_bf16)
// ---------------------------------------------------------------------------
typedef __bf16 bf16_t;
typedef __attribute__((ext_vector_type(8)))  __bf16 v8bf;
typedef __attribute__((ext_vector_type(16))) __bf16 v16bf;
typedef __attribute__((ext_vector_type(8)))  float  v8f;

static __device__ __forceinline__ v8bf ldv8(const bf16_t* p) {
    return *reinterpret_cast<const v8bf*>(p);
}
static __device__ __forceinline__ v16bf cat16(v8bf lo, v8bf hi) {
    return __builtin_shufflevector(lo, hi, 0,1,2,3,4,5,6,7,8,9,10,11,12,13,14,15);
}
static __device__ __forceinline__ v8f wmma_bf16(v16bf a, v16bf b, v8f c) {
    return __builtin_amdgcn_wmma_f32_16x16x32_bf16(false, a, false, b, (short)0, c, false, false);
}

// ---------------------------------------------------------------------------
// Problem constants
// ---------------------------------------------------------------------------
#define BATCH   2
#define SEQ     2048
#define HID     2048
#define NHEADS  16
#define NKV     4
#define HDIM    128
#define MROWS   (BATCH * SEQ)          // 4096
#define NQKV    (HID + 2 * NKV * HDIM) // 3072

// ---------------------------------------------------------------------------
// 1) f32 -> bf16 convert (grid-stride)
// ---------------------------------------------------------------------------
__global__ void cvt_f32_bf16(const float* __restrict__ src, bf16_t* __restrict__ dst, size_t n) {
    for (size_t i = (size_t)blockIdx.x * blockDim.x + threadIdx.x; i < n;
         i += (size_t)gridDim.x * blockDim.x)
        dst[i] = (bf16_t)src[i];
}

// ---------------------------------------------------------------------------
// 2) transpose K x N f32 weight -> N x K bf16  (dst row n is original column n)
// ---------------------------------------------------------------------------
__global__ void transpose_w_bf16(const float* __restrict__ src, bf16_t* __restrict__ dst,
                                 int K, int N) {
    size_t total = (size_t)K * N;
    for (size_t i = (size_t)blockIdx.x * blockDim.x + threadIdx.x; i < total;
         i += (size_t)gridDim.x * blockDim.x) {
        size_t n = i / (size_t)K;
        size_t k = i - n * (size_t)K;
        dst[i] = (bf16_t)src[k * (size_t)N + n];
    }
}

// ---------------------------------------------------------------------------
// 3) GEMM: C[M x N] f32 = A[M x K] bf16 (row major) * Bt[N x K] bf16
//    8 waves/block, wave tile 32x64, block tile 64(M) x 256(N).
// ---------------------------------------------------------------------------
__global__ __launch_bounds__(256) void gemm_bf16_nt(const bf16_t* __restrict__ A,
                                                    const bf16_t* __restrict__ Bt,
                                                    float* __restrict__ C,
                                                    int M, int N, int K) {
    const int lane = threadIdx.x & 31;
    const int wave = threadIdx.x >> 5;
    const int lrow = lane & 15;
    const int lhi  = lane >> 4;
    const int wm = wave & 1, wn = wave >> 1;
    const int mbase = blockIdx.y * 64  + wm * 32;
    const int nbase = blockIdx.x * 256 + wn * 64;

    v8f acc[2][4] = {};

    const bf16_t* Arow0 = A + (size_t)(mbase + lrow) * K;
    const bf16_t* Arow1 = Arow0 + (size_t)16 * K;
    const bf16_t* Brow  = Bt + (size_t)(nbase + lrow) * K;

    for (int k0 = 0; k0 < K; k0 += 32) {
        const int ka = k0 + lhi * 8;           // A frag: elems 0..7 @ka, 8..15 @ka+16
        v16bf a0 = cat16(ldv8(Arow0 + ka), ldv8(Arow0 + ka + 16));
        v16bf a1 = cat16(ldv8(Arow1 + ka), ldv8(Arow1 + ka + 16));
        const int kbm = k0 + lhi * 8;          // B frag: elems 0..7 @kbm, 8..15 @kbm+16
#pragma unroll
        for (int ni = 0; ni < 4; ++ni) {
            const bf16_t* bp = Brow + (size_t)(ni * 16) * K + kbm;
            v16bf b = cat16(ldv8(bp), ldv8(bp + 16));
            acc[0][ni] = wmma_bf16(a0, b, acc[0][ni]);
            acc[1][ni] = wmma_bf16(a1, b, acc[1][ni]);
        }
    }
#pragma unroll
    for (int mi = 0; mi < 2; ++mi)
#pragma unroll
        for (int ni = 0; ni < 4; ++ni) {
            const int col = nbase + ni * 16 + lrow;
#pragma unroll
            for (int r = 0; r < 8; ++r) {
                const int row = mbase + mi * 16 + r + lhi * 8;
                C[(size_t)row * N + col] = acc[mi][ni][r];
            }
        }
}

// ---------------------------------------------------------------------------
// 4) RoPE + split qkv_f32 [4096 x 3072] into:
//    q_bf16 [B,16,S,128] (rope), k_bf16 [B,4,S,128] (rope), vt_bf16 [B,4,128,S]
// ---------------------------------------------------------------------------
__global__ __launch_bounds__(256) void rope_split(const float* __restrict__ qkv,
                                                  bf16_t* __restrict__ q,
                                                  bf16_t* __restrict__ k,
                                                  bf16_t* __restrict__ vt) {
    const int m = blockIdx.x;          // 0..4095
    const int b = m >> 11;
    const int s = m & (SEQ - 1);
    const float* row = qkv + (size_t)m * NQKV;
    const float LOGB = 9.210340371976184f;   // ln(10000)

    for (int t = threadIdx.x; t < 1792; t += 256) {
        if (t < 1024) {                          // q: 16 heads * 64 pairs
            const int h = t >> 6, i = t & 63;
            const float xr = row[h * 128 + 2 * i];
            const float xi = row[h * 128 + 2 * i + 1];
            const float invf = __expf(-((2.0f * (float)i) / 128.0f) * LOGB);
            float sn, cs; __sincosf((float)s * invf, &sn, &cs);
            bf16_t* dst = q + (((size_t)(b * NHEADS + h) * SEQ + s) * HDIM) + 2 * i;
            dst[0] = (bf16_t)(xr * cs - xi * sn);
            dst[1] = (bf16_t)(xr * sn + xi * cs);
        } else if (t < 1280) {                   // k: 4 heads * 64 pairs
            const int j = t - 1024, h = j >> 6, i = j & 63;
            const float xr = row[HID + h * 128 + 2 * i];
            const float xi = row[HID + h * 128 + 2 * i + 1];
            const float invf = __expf(-((2.0f * (float)i) / 128.0f) * LOGB);
            float sn, cs; __sincosf((float)s * invf, &sn, &cs);
            bf16_t* dst = k + (((size_t)(b * NKV + h) * SEQ + s) * HDIM) + 2 * i;
            dst[0] = (bf16_t)(xr * cs - xi * sn);
            dst[1] = (bf16_t)(xr * sn + xi * cs);
        } else {                                 // v: 4 heads * 128 dims -> transposed
            const int j = t - 1280, h = j >> 7, d = j & 127;
            const float xv = row[HID + NKV * HDIM + h * 128 + d];
            vt[(((size_t)(b * NKV + h) * HDIM + d) * SEQ) + s] = (bf16_t)xv;
        }
    }
}

// ---------------------------------------------------------------------------
// 5) Causal flash attention. One wave per 16-query tile, 32-key blocks.
//    Computes S^T = K * Q^T so the softmax P tile is *already* in the WMMA
//    A-fragment layout needed for P*V (no cross-lane transpose).
//    ctx output: bf16 [B, S, 16*128]
// ---------------------------------------------------------------------------
__global__ __launch_bounds__(128) void flash_attn(const bf16_t* __restrict__ Q,
                                                  const bf16_t* __restrict__ K,
                                                  const bf16_t* __restrict__ Vt,
                                                  bf16_t* __restrict__ ctx) {
    const int lane = threadIdx.x & 31;
    const int wave = threadIdx.x >> 5;
    const int lrow = lane & 15;
    const int lhi  = lane >> 4;
    const int qt = blockIdx.x * 4 + wave;   // 0..127 (16-query tile index)
    const int bh = blockIdx.y;              // 0..31
    const int b = bh >> 4, h = bh & 15, kvh = h >> 2;
    const float scale = 0.08838834764831845f;  // 1/sqrt(128)

    const bf16_t* Qp  = Q  + ((size_t)(b * NHEADS + h)  * SEQ) * HDIM;
    const bf16_t* Kp  = K  + ((size_t)(b * NKV   + kvh) * SEQ) * HDIM;
    const bf16_t* Vtp = Vt + ((size_t)(b * NKV   + kvh) * HDIM) * SEQ;

    // Q B-fragments (kept in registers for the whole key loop)
    v16bf qf[4];
    {
        const bf16_t* qrow = Qp + (size_t)(qt * 16 + lrow) * HDIM;
#pragma unroll
        for (int c = 0; c < 4; ++c) {
            const int off = c * 32 + lhi * 8;
            qf[c] = cat16(ldv8(qrow + off), ldv8(qrow + off + 16));
        }
    }

    v8f o[8] = {};
    float m_i = -__builtin_inff();
    float l_i = 0.0f;
    const int q_idx = qt * 16 + lrow;
    const int nkb = (qt * 16 + 47) >> 5;    // ceil((16*qt+16)/32)

    for (int kb = 0; kb < nkb; ++kb) {
        const int keybase = kb * 32;
        const bf16_t* krow0 = Kp + (size_t)(keybase + lrow) * HDIM;
        const bf16_t* krow1 = krow0 + (size_t)16 * HDIM;
        if (kb + 1 < nkb) {   // global_prefetch_b8 for next K and V tiles
            __builtin_prefetch(krow0 + (size_t)32 * HDIM, 0, 0);
            __builtin_prefetch(Vtp + (size_t)lrow * SEQ + keybase + 32, 0, 0);
        }

        // S^T tiles: st0 = keys 0..15 of block, st1 = keys 16..31 (M=key, N=query)
        v8f st0 = {}, st1 = {};
#pragma unroll
        for (int c = 0; c < 4; ++c) {
            const int off = c * 32 + lhi * 8;
            v16bf a0 = cat16(ldv8(krow0 + off), ldv8(krow0 + off + 16));
            v16bf a1 = cat16(ldv8(krow1 + off), ldv8(krow1 + off + 16));
            st0 = wmma_bf16(a0, qf[c], st0);
            st1 = wmma_bf16(a1, qf[c], st1);
        }

        // scale + causal mask + per-lane block max (lane holds query q_idx)
        float p0[8], p1[8];
        float bm = -__builtin_inff();
#pragma unroll
        for (int r = 0; r < 8; ++r) {
            const int key0 = keybase + r + lhi * 8;
            const float s0 = (key0      <= q_idx) ? st0[r] * scale : -__builtin_inff();
            const float s1 = (key0 + 16 <= q_idx) ? st1[r] * scale : -__builtin_inff();
            p0[r] = s0; p1[r] = s1;
            bm = fmaxf(bm, fmaxf(s0, s1));
        }
        bm = fmaxf(bm, __shfl_xor(bm, 16, 32));   // combine key halves (same query)
        const float m_new = fmaxf(m_i, bm);
        const float alpha = __expf(m_i - m_new);

        // P = exp(S - m); P tile lands directly in A-fragment layout
        float rs = 0.0f;
        v16bf pf;
#pragma unroll
        for (int r = 0; r < 8; ++r) {
            const float e0 = __expf(p0[r] - m_new);
            const float e1 = __expf(p1[r] - m_new);
            rs += e0 + e1;
            pf[r]     = (bf16_t)e0;
            pf[r + 8] = (bf16_t)e1;
        }
        rs += __shfl_xor(rs, 16, 32);
        l_i = l_i * alpha + rs;
        m_i = m_new;

        // rescale accumulator: O row r holds query (r + lhi*8); alpha for query j
        // lives in lanes j and j+16 -> src lane = lhi*24 + r
#pragma unroll
        for (int r = 0; r < 8; ++r) {
            const float av = __shfl(alpha, lhi * 24 + r, 32);
#pragma unroll
            for (int c = 0; c < 8; ++c) o[c][r] *= av;
        }

        // O += P * V   (B-fragment from V^T: contiguous along seq)
#pragma unroll
        for (int c = 0; c < 8; ++c) {
            const bf16_t* vp = Vtp + (size_t)(c * 16 + lrow) * SEQ + keybase + lhi * 8;
            v16bf bv = cat16(ldv8(vp), ldv8(vp + 16));
            o[c] = wmma_bf16(pf, bv, o[c]);
        }
    }

    // epilogue: O /= l, write bf16 ctx [B,S,16*128]
#pragma unroll
    for (int r = 0; r < 8; ++r) {
        const float lv  = __shfl(l_i, lhi * 24 + r, 32);
        const float inv = 1.0f / lv;
        const int srow  = qt * 16 + r + lhi * 8;
        bf16_t* crow = ctx + ((size_t)(b * SEQ + srow) * NHEADS + h) * HDIM;
#pragma unroll
        for (int c = 0; c < 8; ++c)
            crow[c * 16 + lrow] = (bf16_t)(o[c][r] * inv);
    }
}

// ---------------------------------------------------------------------------
// Launch orchestration
// ---------------------------------------------------------------------------
extern "C" void kernel_launch(void* const* d_in, const int* in_sizes, int n_in,
                              void* d_out, int out_size, void* d_ws, size_t ws_size,
                              hipStream_t stream) {
    const float* x  = (const float*)d_in[0];
    const float* wq = (const float*)d_in[1];
    const float* wk = (const float*)d_in[2];
    const float* wv = (const float*)d_in[3];
    const float* wo = (const float*)d_in[4];
    float* out = (float*)d_out;

    char* w = (char*)d_ws;
    const size_t MB = 1u << 20;
    bf16_t* xb    = (bf16_t*)(w);               // 16 MB: x bf16 [4096 x 2048]
    bf16_t* wqkvT = (bf16_t*)(w + 16  * MB);    // 12 MB: [3072 x 2048] (col-of-W per row)
    bf16_t* woT   = (bf16_t*)(w + 28  * MB);    //  8 MB: [2048 x 2048]
    float*  qkv   = (float*) (w + 36  * MB);    // 48 MB: [4096 x 3072]
    bf16_t* qb    = (bf16_t*)(w + 84  * MB);    // 16 MB: q rope bf16
    bf16_t* kbuf  = (bf16_t*)(w + 100 * MB);    //  4 MB: k rope bf16
    bf16_t* vtb   = (bf16_t*)(w + 104 * MB);    //  4 MB: v^T bf16
    bf16_t* ctx   = (bf16_t*)(w + 36  * MB);    // 16 MB, aliases qkv (dead after rope)

    // 1) x -> bf16
    cvt_f32_bf16<<<4096, 256, 0, stream>>>(x, xb, (size_t)MROWS * HID);

    // 2) transpose weights -> bf16 (fused qkv rows: [wq | wk | wv])
    transpose_w_bf16<<<4096, 256, 0, stream>>>(wq, wqkvT,                        HID, HID);
    transpose_w_bf16<<<1024, 256, 0, stream>>>(wk, wqkvT + (size_t)2048 * HID,   HID, NKV * HDIM);
    transpose_w_bf16<<<1024, 256, 0, stream>>>(wv, wqkvT + (size_t)2560 * HID,   HID, NKV * HDIM);
    transpose_w_bf16<<<4096, 256, 0, stream>>>(wo, woT,                          HID, HID);

    // 3) qkv = x @ [wq|wk|wv]   (f32 out)
    gemm_bf16_nt<<<dim3(NQKV / 256, MROWS / 64), 256, 0, stream>>>(
        xb, wqkvT, qkv, MROWS, NQKV, HID);

    // 4) RoPE + layout split
    rope_split<<<MROWS, 256, 0, stream>>>(qkv, qb, kbuf, vtb);

    // 5) causal flash attention -> ctx bf16 [B,S,2048]
    flash_attn<<<dim3(SEQ / 16 / 4, BATCH * NHEADS), 128, 0, stream>>>(qb, kbuf, vtb, ctx);

    // 6) out = ctx @ wo  (f32 out)
    gemm_bf16_nt<<<dim3(HID / 256, MROWS / 64), 256, 0, stream>>>(
        ctx, woT, out, MROWS, HID, HID);
}
